// PhiAttention_635655159894
// MI455X (gfx1250) — matrix operand order repaired and checked by reading, not compiled, not run on hardware
//
#include <hip/hip_runtime.h>

// ---------------------------------------------------------------------------
// Problem constants (PhiAttention: B=2, S=2048, H=2560, NH=32, HD=80, RD=32)
// ---------------------------------------------------------------------------
#define B_SZ   2
#define S_LEN  2048
#define H_DIM  2560        // model dim
#define H3     7680        // 3*H
#define NHEAD  32
#define HDIM   80          // head dim
#define SCALE  0.11180339887498948f   // 80^-0.5

typedef __attribute__((ext_vector_type(16))) __bf16 bf16x16;
typedef __attribute__((ext_vector_type(8)))  __bf16 bf16x8;
typedef __attribute__((ext_vector_type(8)))  float  f32x8;

// ---------------------------------------------------------------------------
// bf16 helpers (manual RNE convert)
// ---------------------------------------------------------------------------
__device__ __forceinline__ __bf16 f2bf(float f) {
  unsigned u = __float_as_uint(f);
  u += 0x7FFFu + ((u >> 16) & 1u);
  unsigned short h = (unsigned short)(u >> 16);
  __bf16 r;
  __builtin_memcpy(&r, &h, sizeof(r));
  return r;
}
__device__ __forceinline__ float bf2f(__bf16 b) {
  unsigned short h;
  __builtin_memcpy(&h, &b, sizeof(h));
  return __uint_as_float((unsigned)h << 16);
}
__device__ __forceinline__ bf16x8 bzero8() {
  bf16x8 z;
#pragma unroll
  for (int i = 0; i < 8; ++i) z[i] = f2bf(0.0f);
  return z;
}
__device__ __forceinline__ f32x8 fzero8() {
  f32x8 z = {0.f,0.f,0.f,0.f,0.f,0.f,0.f,0.f};
  return z;
}
__device__ __forceinline__ bf16x16 cat8(bf16x8 lo, bf16x8 hi) {
  return __builtin_shufflevector(lo, hi, 0,1,2,3,4,5,6,7,8,9,10,11,12,13,14,15);
}
__device__ __forceinline__ bf16x16 scale_frag(bf16x16 v, float s) {
  bf16x16 r;
#pragma unroll
  for (int j = 0; j < 16; ++j) r[j] = f2bf(bf2f(v[j]) * s);
  return r;
}
// D = A(16x32 bf16) x B(32x16 bf16) + C(16x16 f32)
__device__ __forceinline__ f32x8 wmma_bf16(bf16x16 a, bf16x16 b, f32x8 c) {
  return __builtin_amdgcn_wmma_f32_16x16x32_bf16(false, a, false, b,
                                                 (short)0, c, false, false);
}

// ---------------------------------------------------------------------------
// fp32 -> bf16 cast (one-time, bandwidth bound)
// ---------------------------------------------------------------------------
__global__ void cvt_f32_to_bf16(const float* __restrict__ src,
                                __bf16* __restrict__ dst, int n) {
  int i = blockIdx.x * blockDim.x + threadIdx.x;
  if (i < n) dst[i] = f2bf(src[i]);
}

// ---------------------------------------------------------------------------
// GEMM: C[M,N] = A[M,K](bf16) * W[N,K]^T(bf16) + bias[N]
// Block = 128 threads = 4 waves; wave tile = 16(M) x 64(N); block = 64x64.
// A fragment (16x32, ISA A-layout): lane m=l&15, half h=l>>4 holds
//   K = h*8+j (j<8) and K = 16+h*8+(j-8) (j>=8)  -> two 16B contiguous loads.
// B fragment (32x16, ISA B-layout): lane n=l&15, half h holds the same K as A
//   -> two 16B contiguous reads of weight row n (since B[k][n] = W[n][k]).
// C/D: n = l&15, m = reg + 8*h.
// k-loop unrolled x2 so the scheduler can overlap one group's global loads
// with the other group's WMMAs; prefetch pulls the next k-step into WGP$.
// ---------------------------------------------------------------------------
template<bool OUT_BF16>
__global__ __launch_bounds__(128)
void gemm_bf16_wmma(const __bf16* __restrict__ A,
                    const __bf16* __restrict__ W,
                    const float*  __restrict__ bias,
                    void* __restrict__ Cout, int K, int N) {
  const int lane = threadIdx.x & 31;
  const int wave = threadIdx.x >> 5;
  const int nlo  = lane & 15;
  const int hh   = lane >> 4;
  const int m0   = blockIdx.y * 64 + wave * 16;
  const int n0   = blockIdx.x * 64;

  f32x8 acc[4];
#pragma unroll
  for (int t = 0; t < 4; ++t) acc[t] = fzero8();

  const __bf16* arow  = A + (size_t)(m0 + nlo) * K;
  const __bf16* wrow0 = W + (size_t)(n0 + nlo) * K;

#pragma unroll 2
  for (int k0 = 0; k0 < K; k0 += 32) {
    if (k0 + 64 < K) {                       // -> global_prefetch_b8
      __builtin_prefetch(arow + k0 + 64, 0, 1);
      __builtin_prefetch(wrow0 + k0 + 64, 0, 1);
    }
    bf16x8 alo = *(const bf16x8*)(arow + k0 + hh * 8);
    bf16x8 ahi = *(const bf16x8*)(arow + k0 + 16 + hh * 8);
    bf16x16 af = cat8(alo, ahi);
#pragma unroll
    for (int t = 0; t < 4; ++t) {
      const __bf16* wrow = wrow0 + (size_t)(t * 16) * K;
      bf16x8 blo = *(const bf16x8*)(wrow + k0 + hh * 8);
      bf16x8 bhi = *(const bf16x8*)(wrow + k0 + 16 + hh * 8);
      acc[t] = wmma_bf16(af, cat8(blo, bhi), acc[t]);
    }
  }

#pragma unroll
  for (int t = 0; t < 4; ++t) {
    const int n = n0 + t * 16 + nlo;
    const float bb = bias[n];
#pragma unroll
    for (int r = 0; r < 8; ++r) {
      const int m = m0 + r + 8 * hh;
      const float v = acc[t][r] + bb;
      if (OUT_BF16) ((__bf16*)Cout)[(size_t)m * N + n] = f2bf(v);
      else          ((float*) Cout)[(size_t)m * N + n] = v;
    }
  }
}

// ---------------------------------------------------------------------------
// NeoX RoPE in-place on the packed qkv buffer [b,s,3H] (q and k slices only).
// ---------------------------------------------------------------------------
__global__ void rope_inplace(__bf16* __restrict__ qkv,
                             const int* __restrict__ pos) {
  const int idx = blockIdx.x * blockDim.x + threadIdx.x;
  // bits: dp[4] | head[5] | t[1] | s[11] | b[1]
  const int dp   = idx & 15;
  const int head = (idx >> 4) & 31;
  const int t    = (idx >> 9) & 1;
  const int s    = (idx >> 10) & (S_LEN - 1);
  const int b    = idx >> 21;
  if (b >= B_SZ) return;

  const float p        = (float)pos[b * S_LEN + s];
  const float inv_freq = __powf(10000.0f, -(float)dp * (1.0f / 16.0f));
  float sn, cs;
  __sincosf(p * inv_freq, &sn, &cs);

  const size_t base = (size_t)(b * S_LEN + s) * H3 + (size_t)t * H_DIM
                      + (size_t)head * HDIM;
  const float x1 = bf2f(qkv[base + dp]);
  const float x2 = bf2f(qkv[base + dp + 16]);
  qkv[base + dp]      = f2bf(x1 * cs - x2 * sn);
  qkv[base + dp + 16] = f2bf(x2 * cs + x1 * sn);
}

// ---------------------------------------------------------------------------
// Flash attention (causal, online softmax), all WMMA bf16.
// Grid (S/64, NH, B), block = 128 = 4 independent wave32's; each wave owns a
// 16-query tile. KV walked in 32-wide blocks. d=80 zero-padded to 96 for QK^T.
// scale folded into the Q fragments.  Row max kept replicated across each
// 16-lane half (shfl_xor reductions never cross the half).  Row SUM is NOT
// shuffled: a 6th PV WMMA against an all-ones B fragment accumulates
// l_new = l_old*corr + rowsum(P) in a C tile under the same correction as the
// output accumulators.  P goes C-layout -> A-layout through a 1KB/wave LDS
// tile; same-wave DS ordering enforced via s_wait_dscnt.
// ---------------------------------------------------------------------------
__global__ __launch_bounds__(128)
void flash_attn_wmma(const __bf16* __restrict__ qkv,
                     __bf16* __restrict__ attn_out) {
  const int lane = threadIdx.x & 31;
  const int wave = threadIdx.x >> 5;
  const int nlo  = lane & 15;
  const int hh   = lane >> 4;
  const int b    = blockIdx.z;
  const int head = blockIdx.y;
  const int q0   = blockIdx.x * 64 + wave * 16;

  __shared__ __bf16 p_lds[4][16][40];   // P tile per wave, padded row stride

  const __bf16* qbase = qkv + (size_t)(b * S_LEN) * H3 + (size_t)head * HDIM;
  const __bf16* kbase = qbase + H_DIM;
  const __bf16* vbase = qbase + 2 * H_DIM;

  // --- Q fragments (A layout), d padded 80 -> 96, pre-scaled by 80^-0.5 ---
  bf16x16 a_q[3];
  {
    const __bf16* qr = qbase + (size_t)(q0 + nlo) * H3;
#pragma unroll
    for (int t = 0; t < 3; ++t) {
      bf16x8 lo = *(const bf16x8*)(qr + t * 32 + hh * 8);   // d = t*32+h*8+j
      bf16x8 hi = (t < 2) ? *(const bf16x8*)(qr + t * 32 + 16 + hh * 8)
                          : bzero8();                        // d >= 80 -> 0
      a_q[t] = scale_frag(cat8(lo, hi), SCALE);
    }
  }

  // all-ones B fragment: P x ones = per-row sum of P, accumulated in accl
  bf16x16 ones;
#pragma unroll
  for (int j = 0; j < 16; ++j) ones[j] = f2bf(1.0f);

  float m_i[8];
  f32x8 acc[5];
  f32x8 accl = fzero8();
#pragma unroll
  for (int r = 0; r < 8; ++r) m_i[r] = -1e30f;
#pragma unroll
  for (int t = 0; t < 5; ++t) acc[t] = fzero8();

  for (int kb = 0; kb < q0 + 16; kb += 32) {
    // ---- S = (Q*scale) K^T : two 16x16 f32 tiles (kv chunks of 16) ----
    f32x8 sc[2];
#pragma unroll
    for (int c = 0; c < 2; ++c) {
      int krow = kb + c * 16 + nlo;
      if (krow > S_LEN - 1) krow = S_LEN - 1;    // masked below; stay in-bounds
      const __bf16* kr = kbase + (size_t)krow * H3;
      f32x8 z = fzero8();
#pragma unroll
      for (int t = 0; t < 3; ++t) {
        bf16x16 bk;
        if (t < 2) {
          bf16x8 lo = *(const bf16x8*)(kr + t * 32 + hh * 8);
          bf16x8 hi = *(const bf16x8*)(kr + t * 32 + 16 + hh * 8);
          bk = cat8(lo, hi);
        } else {                                 // d = 64+h*8+j valid, d >= 80 pad
          bf16x8 lo = *(const bf16x8*)(kr + 64 + hh * 8);
          bk = cat8(lo, bzero8());
        }
        z = wmma_bf16(a_q[t], bk, z);            // EXEC fully converged here
      }
      sc[c] = z;
    }

    // ---- online softmax (per row m = r + 8*hh); only MAX is shuffled ----
#pragma unroll
    for (int r = 0; r < 8; ++r) {
      const int qm = q0 + r + 8 * hh;
      const int kv0 = kb + nlo, kv1 = kb + 16 + nlo;
      const float sv0 = (kv0 <= qm) ? sc[0][r] : -1e30f;
      const float sv1 = (kv1 <= qm) ? sc[1][r] : -1e30f;
      float rmax = fmaxf(sv0, sv1);
#pragma unroll
      for (int msk = 1; msk < 16; msk <<= 1)
        rmax = fmaxf(rmax, __shfl_xor(rmax, msk, 32));
      const float mnew = fmaxf(m_i[r], rmax);
      const float corr = __expf(m_i[r] - mnew);
      m_i[r] = mnew;
      p_lds[wave][r + 8 * hh][nlo]      = f2bf(__expf(sv0 - mnew));
      p_lds[wave][r + 8 * hh][16 + nlo] = f2bf(__expf(sv1 - mnew));
      accl[r] *= corr;
#pragma unroll
      for (int t = 0; t < 5; ++t) acc[t][r] *= corr;
    }

    // same-wave LDS store->load ordering (CDNA5 split counters)
    asm volatile("s_wait_dscnt 0" ::: "memory");

    // ---- P fragment (A layout) from LDS: row = nlo, K = hh*16 + j ----
    bf16x16 a_p;
    {
      const __bf16* pr = &p_lds[wave][nlo][0];
      bf16x8 lo = *(const bf16x8*)(pr + hh * 16);
      bf16x8 hi = *(const bf16x8*)(pr + hh * 16 + 8);
      a_p = cat8(lo, hi);
    }

    // ---- O += P V (5 tiles) and l += P * ones (1 tile) ----
#pragma unroll
    for (int t = 0; t < 5; ++t) {
      bf16x16 bv;
#pragma unroll
      for (int j = 0; j < 16; ++j) {
        int krow = kb + hh * 16 + j;
        if (krow > S_LEN - 1) krow = S_LEN - 1;  // P==0 there, contributes 0
        bv[j] = vbase[(size_t)krow * H3 + t * 16 + nlo];
      }
      acc[t] = wmma_bf16(a_p, bv, acc[t]);
    }
    accl = wmma_bf16(a_p, ones, accl);           // row-sum via tensor op
  }

  // ---- normalize and store [b, q, head*80 + d] as bf16 for out-proj ----
#pragma unroll
  for (int t = 0; t < 5; ++t) {
#pragma unroll
    for (int r = 0; r < 8; ++r) {
      const int qm = q0 + r + 8 * hh;
      const float o = acc[t][r] / accl[r];
      attn_out[(size_t)(b * S_LEN + qm) * H_DIM + (size_t)head * HDIM
               + t * 16 + nlo] = f2bf(o);
    }
  }
}

// ---------------------------------------------------------------------------
// Launch: cast -> QKV GEMM -> RoPE -> flash attention -> out GEMM
// ---------------------------------------------------------------------------
extern "C" void kernel_launch(void* const* d_in, const int* in_sizes, int n_in,
                              void* d_out, int out_size, void* d_ws, size_t ws_size,
                              hipStream_t stream) {
  (void)in_sizes; (void)n_in; (void)out_size; (void)ws_size;
  const float* hidden = (const float*)d_in[0];
  const int*   posids = (const int*)  d_in[1];
  const float* wqkv   = (const float*)d_in[2];
  const float* bqkv   = (const float*)d_in[3];
  const float* wout   = (const float*)d_in[4];
  const float* bout   = (const float*)d_in[5];
  float* out = (float*)d_out;

  char* ws = (char*)d_ws;                       // workspace layout (bytes):
  __bf16* hb    = (__bf16*)(ws);                //  0        : 20,971,520  hidden bf16
  __bf16* wqkvb = (__bf16*)(ws + 20971520);     //            39,321,600  wqkv bf16
  __bf16* woutb = (__bf16*)(ws + 60293120);     //            13,107,200  wout bf16
  __bf16* qkvb  = (__bf16*)(ws + 73400320);     //            62,914,560  qkv bf16
  __bf16* attnb = (__bf16*)(ws + 136314880);    //            20,971,520  attn bf16

  const int nh  = B_SZ * S_LEN * H_DIM;   // 10,485,760
  const int nwq = H3 * H_DIM;             // 19,660,800
  const int nwo = H_DIM * H_DIM;          //  6,553,600
  cvt_f32_to_bf16<<<(nh  + 255) / 256, 256, 0, stream>>>(hidden, hb,    nh);
  cvt_f32_to_bf16<<<(nwq + 255) / 256, 256, 0, stream>>>(wqkv,   wqkvb, nwq);
  cvt_f32_to_bf16<<<(nwo + 255) / 256, 256, 0, stream>>>(wout,   woutb, nwo);

  dim3 g1(H3 / 64, (B_SZ * S_LEN) / 64);        // 120 x 64
  gemm_bf16_wmma<true><<<g1, 128, 0, stream>>>(hb, wqkvb, bqkv, (void*)qkvb,
                                               H_DIM, H3);

  const int nrope = B_SZ * S_LEN * 2 * NHEAD * 16;  // 4,194,304
  rope_inplace<<<nrope / 256, 256, 0, stream>>>(qkvb, posids);

  dim3 g2(S_LEN / 64, NHEAD, B_SZ);             // 32 x 32 x 2
  flash_attn_wmma<<<g2, 128, 0, stream>>>(qkvb, attnb);

  dim3 g3(H_DIM / 64, (B_SZ * S_LEN) / 64);     // 40 x 64
  gemm_bf16_wmma<false><<<g3, 128, 0, stream>>>(attnb, woutb, bout, (void*)out,
                                                H_DIM, H_DIM);
}
